// DentalResPointNet_43568148250762
// MI455X (gfx1250) — matrix operand order repaired and checked by reading, not compiled, not run on hardware
//
#include <hip/hip_runtime.h>

// ---------------------------------------------------------------------------
// DentalResPointNet forward for gfx1250 (MI455X, wave32, WMMA).
// All Linear layers run through v_wmma_f32_16x16x32_f16 with fused epilogues:
//   MODE 0: f32 store (scale/shift)            [shortcut BN, final emb]
//   MODE 1: f16 store, scale/shift + ReLU      [first MLP layer, BN folded]
//   MODE 2: segment-max fused: bias then atomicMax on monotonic float keys
//   MODE 3: f16 store, scale/shift, no ReLU    [FP second layer]
// ---------------------------------------------------------------------------

typedef _Float16 h16;
typedef __attribute__((ext_vector_type(8)))  _Float16 v8h;
typedef __attribute__((ext_vector_type(16))) _Float16 v16h;
typedef __attribute__((ext_vector_type(8)))  float    v8f;

// ---------------- WMMA GEMM ----------------
// A : M x Kp f16 row-major (rows padded to 16, Kp multiple of 32, zero-padded K)
// Bt: N x Kp f16 (weight transposed, zero-padded K)
// out[m,n] = epi( dot(A[m,:],Bt[n,:]) * scale[n] + shift[n] )
template<int MODE>
__global__ __launch_bounds__(256) void k_gemm(
    const h16* __restrict__ A, const h16* __restrict__ Bt,
    const float* __restrict__ scale, const float* __restrict__ shift,
    void* __restrict__ Out, const int* __restrict__ segdst,
    unsigned* __restrict__ keys, int M, int Kp, int N)
{
  const int lane = threadIdx.x & 31;
  const int tile = blockIdx.x * 8 + (threadIdx.x >> 5);
  const int tm   = tile * 16;
  if (tm >= M) return;                      // wave-uniform
  const int tn  = blockIdx.y * 16;
  const int hi  = lane >> 4;                // lanes 16..31
  const int l15 = lane & 15;

  // A fragment: ISA 16x32 f16 layout -> two contiguous 8-half chunks per lane
  const h16* pa = A  + (size_t)(tm + l15) * Kp + (hi ? 8 : 0);
  // B fragment: same layout as A -> two contiguous 8-half chunks per lane
  const h16* pb = Bt + (size_t)(tn + l15) * Kp + (hi ? 8 : 0);

  v8f acc = {};
  for (int kk = 0; kk < Kp; kk += 32) {
    v8h a0 = *(const v8h*)(pa + kk);        // halves 0..7
    v8h a1 = *(const v8h*)(pa + kk + 16);   // halves 8..15
    v16h a;
#pragma unroll
    for (int i = 0; i < 8; ++i) { a[i] = a0[i]; a[i + 8] = a1[i]; }
    v8h b0 = *(const v8h*)(pb + kk);
    v8h b1 = *(const v8h*)(pb + kk + 16);
    v16h b;
#pragma unroll
    for (int i = 0; i < 8; ++i) { b[i] = b0[i]; b[i + 8] = b1[i]; }
    acc = __builtin_amdgcn_wmma_f32_16x16x32_f16(false, a, false, b,
                                                 (short)0, acc, false, false);
  }

  const int   n = tn + l15;
  const float s = scale[n], t = shift[n];
#pragma unroll
  for (int v = 0; v < 8; ++v) {
    const int m = tm + v + (hi ? 8 : 0);    // C/D layout: VGPR v -> row v(+8)
    if (m < M) {
      float val = acc[v] * s + t;
      if (MODE == 1) val = fmaxf(val, 0.0f);
      if (MODE == 0) {
        ((float*)Out)[(size_t)m * N + n] = val;
      } else if (MODE == 1 || MODE == 3) {
        ((h16*)Out)[(size_t)m * N + n] = (h16)val;
      } else { // MODE 2: fused segment_max (order-invariant monotonic key)
        unsigned b32 = __float_as_uint(val);
        unsigned key = (b32 & 0x80000000u) ? ~b32 : (b32 | 0x80000000u);
        atomicMax(&keys[(size_t)segdst[m] * N + n], key);
      }
    }
  }
}

// ---------------- prep kernels ----------------
__global__ void k_wtrans(const float* __restrict__ W, h16* __restrict__ Wt,
                         int K, int N, int Kp) {
  int id = blockIdx.x * blockDim.x + threadIdx.x;
  if (id >= N * Kp) return;
  int n = id / Kp, k = id % Kp;
  Wt[id] = (k < K) ? (h16)W[(size_t)k * N + n] : (h16)0.f;
}
__global__ void k_bnss(const float* b, const float* g, const float* be,
                       const float* m, const float* v,
                       float* scale, float* shift, int N) {
  int n = threadIdx.x; if (n >= N) return;
  float s = g[n] * rsqrtf(v[n] + 1e-5f);
  scale[n] = s; shift[n] = (b[n] - m[n]) * s + be[n];
}
__global__ void k_biass(const float* b, float* scale, float* shift, int N) {
  int n = threadIdx.x; if (n >= N) return;
  scale[n] = 1.0f; shift[n] = b[n];
}
__global__ void k_fill0(unsigned* p, size_t n) {
  size_t id = (size_t)blockIdx.x * blockDim.x + threadIdx.x;
  if (id < n) p[id] = 0u;
}

// ---------------- feature assembly ----------------
__global__ void k_pos16(const float* __restrict__ pos, h16* __restrict__ out) {
  int r = blockIdx.x, c = threadIdx.x; // 32 threads
  out[(size_t)r * 32 + c] = (c < 3) ? (h16)pos[r * 3 + c] : (h16)0.f;
}
__global__ void k_edge_pos(const float* __restrict__ pos, const int* __restrict__ src,
                           const int* __restrict__ dst, h16* __restrict__ feat) {
  int e = blockIdx.x, c = threadIdx.x; // 32 threads, Kp=32
  int s = src[e];
  h16 v = (h16)0.f;
  if (c < 3) v = (h16)pos[s * 3 + c];
  else if (c < 6) { int k = c - 3, d = dst[e]; v = (h16)(pos[s*3+k] - pos[d*3+k]); }
  feat[(size_t)e * 32 + c] = v;
}
__global__ void k_edge_f16(const h16* __restrict__ x, int Cx,
                           const float* __restrict__ pos,
                           const int* __restrict__ src, const int* __restrict__ dst,
                           h16* __restrict__ feat, int Kp) {
  int e = blockIdx.x, c = threadIdx.x;
  int s = src[e];
  h16 v;
  if (c < Cx)            v = x[(size_t)s * Cx + c];
  else if (c < Cx + 3) { int k = c - Cx, d = dst[e]; v = (h16)(pos[s*3+k] - pos[d*3+k]); }
  else                   v = (h16)0.f;
  feat[(size_t)e * Kp + c] = v;
}
__global__ void k_gather_pos(const float* __restrict__ pos, const int* __restrict__ idx,
                             float* __restrict__ out, int R) {
  int id = blockIdx.x * blockDim.x + threadIdx.x;
  if (id >= R * 3) return;
  int r = id / 3, k = id % 3;
  out[id] = pos[(size_t)idx[r] * 3 + k];
}
// agg decode + shortcut + ReLU (+ optional row gather), f16 out
__global__ void k_combine(const unsigned* __restrict__ keys, const float* __restrict__ sc,
                          const h16* __restrict__ xin, const int* __restrict__ gidx,
                          h16* __restrict__ out, int C) {
  int r = blockIdx.x, c = threadIdx.x;
  int i = gidx ? gidx[r] : r;
  unsigned k = keys[(size_t)i * C + c];
  float a = 0.f;
  if (k) {
    unsigned b = (k & 0x80000000u) ? (k ^ 0x80000000u) : ~k;
    float f = __uint_as_float(b);
    a = __builtin_isfinite(f) ? f : 0.f;
  }
  float s = sc ? sc[(size_t)i * C + c] : (float)xin[(size_t)i * C + c];
  out[(size_t)r * C + c] = (h16)fmaxf(a + s, 0.f);
}
// 3-NN inverse-distance interp, writes cols [0,C) of padded A buffer
__global__ void k_interp(const h16* __restrict__ xs, const float* __restrict__ ps,
                         const float* __restrict__ pt, const int* __restrict__ nn,
                         h16* __restrict__ A, int C, int Kp) {
  int t = blockIdx.x, c = threadIdx.x;
  int i0 = nn[t*3], i1 = nn[t*3+1], i2 = nn[t*3+2];
  float px = pt[t*3], py = pt[t*3+1], pz = pt[t*3+2];
  auto w = [&](int i) {
    float dx = px - ps[i*3], dy = py - ps[i*3+1], dz = pz - ps[i*3+2];
    return 1.f / (dx*dx + dy*dy + dz*dz + 1e-16f);
  };
  float w0 = w(i0), w1 = w(i1), w2 = w(i2);
  float val = (w0 * (float)xs[(size_t)i0*C + c] + w1 * (float)xs[(size_t)i1*C + c]
             + w2 * (float)xs[(size_t)i2*C + c]) / (w0 + w1 + w2);
  A[(size_t)t * Kp + c] = (h16)val;
}
__global__ void k_tail_f16(const h16* __restrict__ x, h16* __restrict__ A,
                           int Cx, int Ci, int Kp) {
  int t = blockIdx.x, c = threadIdx.x;
  A[(size_t)t * Kp + Ci + c] = (c < Cx) ? x[(size_t)t * Cx + c] : (h16)0.f;
}
__global__ void k_tail_pos(const float* __restrict__ pos, h16* __restrict__ A,
                           int Ci, int Kp) {
  int t = blockIdx.x, c = threadIdx.x;
  A[(size_t)t * Kp + Ci + c] = (c < 3) ? (h16)pos[t * 3 + c] : (h16)0.f;
}

// ---------------- ArcFace head ----------------
__global__ void k_labmin_init(int* p) { *p = 0x7fffffff; }
__global__ void k_labmin(const int* __restrict__ label, int n, int* out) {
  int i = blockIdx.x * blockDim.x + threadIdx.x;
  if (i < n) atomicMin(out, label[i]);
}
__global__ void k_arc_wn(const float* __restrict__ W, float* __restrict__ wn) {
  int c = threadIdx.x; if (c >= 3) return;
  float ss = 0.f;
  for (int i = 0; i < 128; ++i) { float x = W[c*128+i]; ss += x*x; }
  float inv = rsqrtf(ss);
  for (int i = 0; i < 128; ++i) wn[c*128+i] = W[c*128+i] * inv;
}
__global__ void k_arc_out(const float* __restrict__ emb, const float* __restrict__ wn,
                          const int* __restrict__ label, const int* __restrict__ labmin,
                          float* __restrict__ out, int n0) {
  int r = blockIdx.x * blockDim.x + threadIdx.x;
  if (r >= n0) return;
  const float* e = emb + (size_t)r * 128;
  float ss = 0.f, d0 = 0.f, d1 = 0.f, d2 = 0.f;
  for (int i = 0; i < 128; ++i) {
    float x = e[i];
    ss += x * x;
    d0 += x * wn[i]; d1 += x * wn[128 + i]; d2 += x * wn[256 + i];
  }
  float inv = rsqrtf(ss);
  int lab = label[r] - ((*labmin >= 1) ? 1 : 0);
  float cosv[3] = { d0 * inv, d1 * inv, d2 * inv };
#pragma unroll
  for (int c = 0; c < 3; ++c) {
    float cc = fminf(fmaxf(cosv[c], -1.f + 1e-7f), 1.f - 1e-7f);
    float tg = cosf(acosf(cc) + 0.5f);
    out[(size_t)r * 3 + c] = 30.0f * ((c == lab) ? tg : cosv[c]);
  }
}

// ---------------- host orchestration ----------------
struct Lin { h16* Wt; float* scale; float* shift; int K, Kp, N; };

extern "C" void kernel_launch(void* const* d_in, const int* in_sizes, int n_in,
                              void* d_out, int out_size, void* d_ws, size_t ws_size,
                              hipStream_t stream)
{
  (void)n_in; (void)out_size; (void)ws_size;
  // pytree flatten: top-level in setup_inputs() order; params dict depth-first,
  // sorted keys ('mlp' before 'sc'; W1 W2 b1 b2 be1 g1 m1 v1; Ws bes bs gs ms vs)
  enum {
    IN_POS = 0,
    P_ARCW = 1,
    P_E1_W1 = 2,  P_E1_WS = 10,
    P_E1B_W1 = 16,
    P_E2_W1 = 24, P_E2_WS = 32,
    P_E2B_W1 = 38,
    P_E3_W1 = 46, P_E3_WS = 54,
    P_FP1_W1 = 60, P_FP2_W1 = 68, P_FP3_W1 = 76,
    IN_LABEL = 84, IN_IDX1, IN_IDX2, IN_IDX3,
    IN_E1S, IN_E1D, IN_E1BS, IN_E1BD,
    IN_E2S, IN_E2D, IN_E2BS, IN_E2BD,
    IN_E3S, IN_E3D, IN_KNN3, IN_KNN2, IN_KNN1
  };
#define FP_(i) ((const float*)d_in[i])
#define IP_(i) ((const int*)d_in[i])

  const int n0 = in_sizes[IN_POS] / 3;
  const int n1 = in_sizes[IN_IDX1], n2 = in_sizes[IN_IDX2], n3 = in_sizes[IN_IDX3];
  const int E1 = in_sizes[IN_E1S], E1b = in_sizes[IN_E1BS];
  const int E2 = in_sizes[IN_E2S], E2b = in_sizes[IN_E2BS], E3 = in_sizes[IN_E3S];

  const float* pos   = FP_(IN_POS);
  const int* idx1 = IP_(IN_IDX1), *idx2 = IP_(IN_IDX2), *idx3 = IP_(IN_IDX3);

  char* wsp = (char*)d_ws;
  size_t off = 0;
  auto alloc = [&](size_t bytes) -> void* {
    off = (off + 255) & ~(size_t)255;
    void* p = wsp + off; off += bytes; return p;
  };

  // ---- prep folded linears (tiny kernels, done every call: deterministic) ----
  auto prep = [&](const float* W, const float* b, const float* g, const float* be,
                  const float* m, const float* v, int K, int N) -> Lin {
    Lin L; L.K = K; L.N = N; L.Kp = (K + 31) & ~31;
    L.Wt = (h16*)alloc((size_t)N * L.Kp * 2);
    L.scale = (float*)alloc((size_t)N * 4);
    L.shift = (float*)alloc((size_t)N * 4);
    int tot = N * L.Kp;
    k_wtrans<<<(tot + 255) / 256, 256, 0, stream>>>(W, L.Wt, K, N, L.Kp);
    if (g) k_bnss<<<1, N, 0, stream>>>(b, g, be, m, v, L.scale, L.shift, N);
    else   k_biass<<<1, N, 0, stream>>>(b, L.scale, L.shift, N);
    return L;
  };
  auto prepM1 = [&](int b, int K, int N) { // mlp first layer: BN(W1x+b1)
    return prep(FP_(b), FP_(b+2), FP_(b+5), FP_(b+4), FP_(b+6), FP_(b+7), K, N); };
  auto prepM2 = [&](int b, int K, int N) { // mlp second layer: W2x+b2
    return prep(FP_(b+1), FP_(b+3), nullptr, nullptr, nullptr, nullptr, K, N); };
  auto prepSC = [&](int b, int K, int N) { // shortcut: BN(Ws x + bs)
    return prep(FP_(b), FP_(b+2), FP_(b+3), FP_(b+1), FP_(b+4), FP_(b+5), K, N); };

  Lin e1a  = prepM1(P_E1_W1,   6,  64), e1b_ = prepM2(P_E1_W1,  64,  64);
  Lin e1s  = prepSC(P_E1_WS,   3,  64);
  Lin eb1a = prepM1(P_E1B_W1, 67,  64), eb1b = prepM2(P_E1B_W1, 64,  64);
  Lin e2a  = prepM1(P_E2_W1,  67, 128), e2b_ = prepM2(P_E2_W1, 128, 128);
  Lin e2s  = prepSC(P_E2_WS,  64, 128);
  Lin eb2a = prepM1(P_E2B_W1,131, 128), eb2b = prepM2(P_E2B_W1,128, 128);
  Lin e3a  = prepM1(P_E3_W1, 131, 256), e3b_ = prepM2(P_E3_W1, 256, 256);
  Lin e3s  = prepSC(P_E3_WS, 128, 256);
  Lin f3a  = prepM1(P_FP3_W1,384, 256), f3b  = prepM2(P_FP3_W1,256, 128);
  Lin f2a  = prepM1(P_FP2_W1,192, 128), f2b  = prepM2(P_FP2_W1,128, 128);
  Lin f1a  = prepM1(P_FP1_W1,131, 128), f1b  = prepM2(P_FP1_W1,128, 128);

  // ---- persistent buffers ----
  h16*   pos16 = (h16*)alloc((size_t)n0 * 32 * 2);
  float* pos1  = (float*)alloc((size_t)n1 * 3 * 4);
  float* pos2  = (float*)alloc((size_t)n2 * 3 * 4);
  float* pos3  = (float*)alloc((size_t)n3 * 3 * 4);
  h16*   x1    = (h16*)alloc((size_t)n1 * 64 * 2);
  h16*   x1b   = (h16*)alloc((size_t)n1 * 64 * 2);
  h16*   x2    = (h16*)alloc((size_t)n2 * 128 * 2);
  h16*   x2b   = (h16*)alloc((size_t)n2 * 128 * 2);
  h16*   x3    = (h16*)alloc((size_t)n3 * 256 * 2);
  h16*   h2    = (h16*)alloc((size_t)n2 * 128 * 2);
  h16*   h1    = (h16*)alloc((size_t)n1 * 128 * 2);
  float* emb   = (float*)alloc((size_t)n0 * 128 * 4);
  float* wn    = (float*)alloc(384 * 4);
  int*   labmin= (int*)alloc(4);

  auto gemm = [&](int mode, const h16* A, const Lin& L, int M, void* out,
                  const int* dstp, unsigned* keys) {
    int tiles = ((M + 15) & ~15) / 16;
    dim3 grid((tiles + 7) / 8, L.N / 16);
    if (mode == 0)      k_gemm<0><<<grid, 256, 0, stream>>>(A, L.Wt, L.scale, L.shift, out, dstp, keys, M, L.Kp, L.N);
    else if (mode == 1) k_gemm<1><<<grid, 256, 0, stream>>>(A, L.Wt, L.scale, L.shift, out, dstp, keys, M, L.Kp, L.N);
    else if (mode == 2) k_gemm<2><<<grid, 256, 0, stream>>>(A, L.Wt, L.scale, L.shift, out, dstp, keys, M, L.Kp, L.N);
    else                k_gemm<3><<<grid, 256, 0, stream>>>(A, L.Wt, L.scale, L.shift, out, dstp, keys, M, L.Kp, L.N);
  };

  // message-passing block: gather edges -> GEMM1(relu) -> GEMM2(fused segmax)
  // -> shortcut GEMM (or identity) -> combine(+gather)
  auto run_block = [&](const h16* x, int Cx, const float* posl,
                       const int* src, const int* dst, int E, int nseg,
                       const Lin& L1, const Lin& L2, const Lin* Lsc, const h16* scA,
                       const int* gidx, h16* out, int Rout) {
    size_t mark = off;
    int Ep = (E + 15) & ~15;
    h16* feat = (h16*)alloc((size_t)Ep * L1.Kp * 2);
    h16* hbuf = (h16*)alloc((size_t)Ep * L1.N * 2);
    unsigned* keys = (unsigned*)alloc((size_t)nseg * L2.N * 4);
    float* sc = Lsc ? (float*)alloc((size_t)nseg * L2.N * 4) : nullptr;
    size_t nk = (size_t)nseg * L2.N;
    k_fill0<<<(unsigned)((nk + 255) / 256), 256, 0, stream>>>(keys, nk);
    if (E > 0) {
      if (x) k_edge_f16<<<E, L1.Kp, 0, stream>>>(x, Cx, posl, src, dst, feat, L1.Kp);
      else   k_edge_pos<<<E, 32, 0, stream>>>(posl, src, dst, feat);
      gemm(1, feat, L1, E, hbuf, nullptr, nullptr);
      gemm(2, hbuf, L2, E, nullptr, dst, keys);
    }
    if (Lsc) gemm(0, scA, *Lsc, nseg, sc, nullptr, nullptr);
    k_combine<<<Rout, L2.N, 0, stream>>>(keys, sc, x, gidx, out, L2.N);
    off = mark;
  };

  // feature propagation: interp + concat -> GEMM1(relu) -> GEMM2
  auto run_fp = [&](const h16* xs, int Ci, const float* ps, const float* pt,
                    const int* nn, int T, const h16* xtail, int Cx,
                    const float* ptail, const Lin& L1, const Lin& L2,
                    int mode2, void* out) {
    size_t mark = off;
    int Tp = (T + 15) & ~15;
    h16* Abuf = (h16*)alloc((size_t)Tp * L1.Kp * 2);
    h16* hbuf = (h16*)alloc((size_t)Tp * L1.N * 2);
    k_interp<<<T, Ci, 0, stream>>>(xs, ps, pt, nn, Abuf, Ci, L1.Kp);
    int tailw = L1.Kp - Ci;
    if (xtail) k_tail_f16<<<T, tailw, 0, stream>>>(xtail, Abuf, Cx, Ci, L1.Kp);
    else       k_tail_pos<<<T, tailw, 0, stream>>>(ptail, Abuf, Ci, L1.Kp);
    gemm(1, Abuf, L1, T, hbuf, nullptr, nullptr);
    gemm(mode2, hbuf, L2, T, out, nullptr, nullptr);
    off = mark;
  };

  // ---- pipeline ----
  k_pos16<<<n0, 32, 0, stream>>>(pos, pos16);
  k_gather_pos<<<(n1 * 3 + 255) / 256, 256, 0, stream>>>(pos,  idx1, pos1, n1);
  k_gather_pos<<<(n2 * 3 + 255) / 256, 256, 0, stream>>>(pos1, idx2, pos2, n2);
  k_gather_pos<<<(n3 * 3 + 255) / 256, 256, 0, stream>>>(pos2, idx3, pos3, n3);

  run_block(nullptr, 0, pos, IP_(IN_E1S),  IP_(IN_E1D),  E1,  n0, e1a,  e1b_, &e1s, pos16, idx1, x1,  n1); // enc1 [idx1]
  run_block(x1,  64, pos1,  IP_(IN_E1BS), IP_(IN_E1BD), E1b, n1, eb1a, eb1b, nullptr, nullptr, nullptr, x1b, n1); // enc1b
  run_block(x1b, 64, pos1,  IP_(IN_E2S),  IP_(IN_E2D),  E2,  n1, e2a,  e2b_, &e2s, x1b,  idx2, x2,  n2); // enc2 [idx2]
  run_block(x2, 128, pos2,  IP_(IN_E2BS), IP_(IN_E2BD), E2b, n2, eb2a, eb2b, nullptr, nullptr, nullptr, x2b, n2); // enc2b
  run_block(x2b,128, pos2,  IP_(IN_E3S),  IP_(IN_E3D),  E3,  n2, e3a,  e3b_, &e3s, x2b,  idx3, x3,  n3); // enc3 [idx3]

  run_fp(x3, 256, pos3, pos2, IP_(IN_KNN3), n2, x2b, 128, nullptr, f3a, f3b, 3, h2);  // fp3
  run_fp(h2, 128, pos2, pos1, IP_(IN_KNN2), n1, x1b,  64, nullptr, f2a, f2b, 3, h1);  // fp2
  run_fp(h1, 128, pos1, pos,  IP_(IN_KNN1), n0, nullptr, 3, pos,   f1a, f1b, 0, emb); // fp1 -> f32 emb

  // ---- ArcFace head ----
  k_labmin_init<<<1, 1, 0, stream>>>(labmin);
  k_labmin<<<(n0 + 255) / 256, 256, 0, stream>>>(IP_(IN_LABEL), n0, labmin);
  k_arc_wn<<<1, 3, 0, stream>>>(FP_(P_ARCW), wn);
  k_arc_out<<<(n0 + 255) / 256, 256, 0, stream>>>(emb, wn, IP_(IN_LABEL), labmin,
                                                  (float*)d_out, n0);
#undef FP_
#undef IP_
}
